// SimpleMoE_80204219286163
// MI455X (gfx1250) — matrix operand order repaired and checked by reading, not compiled, not run on hardware
//
#include <hip/hip_runtime.h>
#include <cmath>
#include <cstdint>

#define TOKENS 4096
#define DIM 1024
#define HID 2048
#define NEXP 4

typedef __attribute__((ext_vector_type(16))) __bf16 v16bf;
typedef __attribute__((ext_vector_type(8)))  __bf16 v8bf;
typedef __attribute__((ext_vector_type(8)))  float  v8f;
typedef __attribute__((ext_vector_type(4)))  int    v4i;

typedef __attribute__((address_space(1))) v4i* as1_v4i_ptr;
typedef __attribute__((address_space(3))) v4i* as3_v4i_ptr;

union FragAB { v16bf v; v8bf h[2]; };

#if __has_builtin(__builtin_amdgcn_global_load_async_to_lds_b128)
#define USE_ASYNC 1
#else
#define USE_ASYNC 0
#endif

#if __has_builtin(__builtin_amdgcn_s_wait_asynccnt)
#define WAIT_ASYNC0() __builtin_amdgcn_s_wait_asynccnt(0)
#else
#define WAIT_ASYNC0() asm volatile("s_wait_asynccnt 0" ::: "memory")
#endif

__device__ __forceinline__ v8f wmma_bf16(v16bf a, v16bf b, v8f c) {
  // D = A(16x32 bf16) * B(32x16 bf16) + C(16x16 f32)
  return __builtin_amdgcn_wmma_f32_16x16x32_bf16(
      /*neg_a=*/false, a, /*neg_b=*/false, b,
      /*c_mod=*/(short)0, c, /*reuse_a=*/false, /*reuse_b=*/false);
}

__device__ __forceinline__ float gelu_exact(float v) {
  return 0.5f * v * (1.0f + erff(v * 0.70710678118654752f));
}

// ---------------------------------------------------------------------------
// f32 -> bf16 elementwise convert (4 elements / thread)
// ---------------------------------------------------------------------------
__global__ __launch_bounds__(256) void cvt_fp32_bf16(const float* __restrict__ in,
                                                     __bf16* __restrict__ out) {
  const size_t i = (size_t)blockIdx.x * blockDim.x + threadIdx.x;
  float4 v = *(const float4*)(in + i * 4);
  union { __bf16 b[4]; uint2 u; } t;
  t.b[0] = (__bf16)v.x; t.b[1] = (__bf16)v.y;
  t.b[2] = (__bf16)v.z; t.b[3] = (__bf16)v.w;
  *(uint2*)(out + i * 4) = t.u;
}

// ---------------------------------------------------------------------------
// f32 [R][C] -> bf16 [C][R] transpose+convert, 32x32 LDS tiles, z = expert.
// ---------------------------------------------------------------------------
__global__ __launch_bounds__(256) void transpose_cvt(const float* __restrict__ in,
                                                     __bf16* __restrict__ out,
                                                     int R, int C) {
  __shared__ float t[32][33];
  const size_t plane = (size_t)R * C;
  const float* ine = in + (size_t)blockIdx.z * plane;
  __bf16* oute = out + (size_t)blockIdx.z * plane;
  const int tx = threadIdx.x & 31, ty = threadIdx.x >> 5;
  const int r0 = blockIdx.y * 32, c0 = blockIdx.x * 32;
#pragma unroll
  for (int i = 0; i < 4; ++i) {
    int r = ty + i * 8;
    t[r][tx] = ine[(size_t)(r0 + r) * C + c0 + tx];
  }
  __syncthreads();
#pragma unroll
  for (int i = 0; i < 4; ++i) {
    int oc = ty + i * 8;
    oute[(size_t)(c0 + oc) * R + r0 + tx] = (__bf16)t[tx][oc];
  }
}

// ---------------------------------------------------------------------------
// Router: weights = softmax(x @ Wr + br). One wave per token.
// ---------------------------------------------------------------------------
__global__ __launch_bounds__(256) void moe_router(const float* __restrict__ x,
                                                  const float* __restrict__ Wr,
                                                  const float* __restrict__ br,
                                                  float* __restrict__ wts) {
  const int token = (blockIdx.x * blockDim.x + threadIdx.x) >> 5;
  const int lane  = threadIdx.x & 31;
  const float* xr = x + (size_t)token * DIM;
  float a0 = 0.f, a1 = 0.f, a2 = 0.f, a3 = 0.f;
  for (int d = lane; d < DIM; d += 32) {
    float xv = xr[d];
    float4 w = *(const float4*)(Wr + d * 4);  // Wr is [DIM][4] row-major
    a0 = fmaf(xv, w.x, a0);
    a1 = fmaf(xv, w.y, a1);
    a2 = fmaf(xv, w.z, a2);
    a3 = fmaf(xv, w.w, a3);
  }
#pragma unroll
  for (int off = 16; off > 0; off >>= 1) {
    a0 += __shfl_down(a0, off, 32);
    a1 += __shfl_down(a1, off, 32);
    a2 += __shfl_down(a2, off, 32);
    a3 += __shfl_down(a3, off, 32);
  }
  if (lane == 0) {
    a0 += br[0]; a1 += br[1]; a2 += br[2]; a3 += br[3];
    float mx = fmaxf(fmaxf(a0, a1), fmaxf(a2, a3));
    float e0 = expf(a0 - mx), e1 = expf(a1 - mx);
    float e2 = expf(a2 - mx), e3 = expf(a3 - mx);
    float inv = 1.0f / (e0 + e1 + e2 + e3);
    float4 o = {e0 * inv, e1 * inv, e2 * inv, e3 * inv};
    *(float4*)(wts + (size_t)token * 4) = o;
  }
}

// ---------------------------------------------------------------------------
// Unified bf16 WMMA GEMM: C[M,N] = A[M,K] @ BT[N,K]^T
//   EPI==0: H = gelu(C + bias) -> bf16 (N stride HID)
//   EPI==1: out (+)= wts[m][expert] * (C + bias)  (N stride DIM, e==0 writes)
// 128x128 block tile, BK=32, 8 waves (2x4), each wave 64x32 -> 4x2 WMMA tiles.
// LDS tiles double-buffered, filled via global_load_async_to_lds_b128.
// ---------------------------------------------------------------------------
template <int K, int EPI>
__global__ __launch_bounds__(256) void moe_gemm(const __bf16* __restrict__ A,
                                                const __bf16* __restrict__ BT,
                                                const float* __restrict__ bias,
                                                const float* __restrict__ wts,
                                                float* __restrict__ outF,
                                                __bf16* __restrict__ outH,
                                                int expert) {
  constexpr int LDO = (EPI == 0) ? HID : DIM;
  __shared__ __align__(16) __bf16 Als[2][128][40];  // [m][k], padded rows
  __shared__ __align__(16) __bf16 Bls[2][128][40];  // [n][k]

  const int tid = threadIdx.x;
  const int m0 = blockIdx.x * 128;
  const int n0 = blockIdx.y * 128;
  const int wave = tid >> 5, lane = tid & 31;
  const int wm = wave >> 2, wn = wave & 3;
  const int lane16 = lane & 15, khalf = lane >> 4;

  v8f acc[4][2] = {};

  // Copy a 128-row x 32-bf16 (64B) tile into LDS; wave w owns rows w*16..w*16+15,
  // each instruction moves 8 rows x 64B (16B per lane).
  auto copy_tile = [&](const __bf16* src, __bf16 (*dst)[40]) {
#pragma unroll
    for (int i = 0; i < 2; ++i) {
      int row = wave * 16 + i * 8 + (lane >> 2);
      int cc  = (lane & 3) * 8;  // bf16 column
      const __bf16* g = src + (size_t)row * K + cc;
      __bf16* l = &dst[row][cc];
#if USE_ASYNC
      __builtin_amdgcn_global_load_async_to_lds_b128(
          (as1_v4i_ptr)(uintptr_t)g,
          (as3_v4i_ptr)(uint32_t)(uintptr_t)l,
          /*offset=*/0, /*cpol=*/0);
#else
      *(uint4*)l = *(const uint4*)g;
#endif
    }
  };

  const __bf16* Asrc = A + (size_t)m0 * K;
  const __bf16* Bsrc = BT + (size_t)n0 * K;

  copy_tile(Asrc, Als[0]);
  copy_tile(Bsrc, Bls[0]);
#if USE_ASYNC
  WAIT_ASYNC0();
#endif
  __syncthreads();

  constexpr int nk = K / 32;
  for (int kt = 0; kt < nk; ++kt) {
    const int buf = kt & 1;
    if (kt + 1 < nk) {
      copy_tile(Asrc + (kt + 1) * 32, Als[buf ^ 1]);
      copy_tile(Bsrc + (kt + 1) * 32, Bls[buf ^ 1]);
    }

    FragAB afr[4], bfr[2];
#pragma unroll
    for (int tm = 0; tm < 4; ++tm) {
      const __bf16* p = &Als[buf][wm * 64 + tm * 16 + lane16][khalf * 8];
      afr[tm].h[0] = *(const v8bf*)p;         // K = khalf*8 .. +7
      afr[tm].h[1] = *(const v8bf*)(p + 16);  // K = 16+khalf*8 .. +7
    }
#pragma unroll
    for (int tn = 0; tn < 2; ++tn) {
      const __bf16* p = &Bls[buf][wn * 32 + tn * 16 + lane16][khalf * 8];
      bfr[tn].h[0] = *(const v8bf*)p;         // K = khalf*8 .. +7
      bfr[tn].h[1] = *(const v8bf*)(p + 16);  // K = 16+khalf*8 .. +7
    }
#pragma unroll
    for (int tm = 0; tm < 4; ++tm)
#pragma unroll
      for (int tn = 0; tn < 2; ++tn)
        acc[tm][tn] = wmma_bf16(afr[tm].v, bfr[tn].v, acc[tm][tn]);

#if USE_ASYNC
    if (kt + 1 < nk) WAIT_ASYNC0();
#endif
    __syncthreads();
  }

  // epilogue
#pragma unroll
  for (int tm = 0; tm < 4; ++tm) {
#pragma unroll
    for (int tn = 0; tn < 2; ++tn) {
      int n = n0 + wn * 32 + tn * 16 + lane16;
      float bv = bias[n];
#pragma unroll
      for (int r = 0; r < 8; ++r) {
        int m = m0 + wm * 64 + tm * 16 + khalf * 8 + r;
        float v = acc[tm][tn][r] + bv;
        if constexpr (EPI == 0) {
          outH[(size_t)m * LDO + n] = (__bf16)gelu_exact(v);
        } else {
          float wgt = wts[(size_t)m * NEXP + expert];
          size_t idx = (size_t)m * LDO + n;
          float prev = (expert == 0) ? 0.0f : outF[idx];
          outF[idx] = fmaf(wgt, v, prev);
        }
      }
    }
  }
}

// ---------------------------------------------------------------------------
extern "C" void kernel_launch(void* const* d_in, const int* in_sizes, int n_in,
                              void* d_out, int out_size, void* d_ws, size_t ws_size,
                              hipStream_t stream) {
  const float* x  = (const float*)d_in[0];  // [2,2048,1024]
  const float* W1 = (const float*)d_in[1];  // [4,1024,2048]
  const float* b1 = (const float*)d_in[2];  // [4,2048]
  const float* W2 = (const float*)d_in[3];  // [4,2048,1024]
  const float* b2 = (const float*)d_in[4];  // [4,1024]
  const float* Wr = (const float*)d_in[5];  // [1024,4]
  const float* br = (const float*)d_in[6];  // [4]
  float* out = (float*)d_out;               // [2,2048,1024]

  // workspace layout (56.06 MB):
  //   [0, 64KB)          router weights f32 [4096][4]
  //   [64KB, +8MB)       x   bf16 [4096][1024]
  //   [+8MB, +24MB)      H   bf16 [4096][2048]
  //   [+24MB, +40MB)     W1T bf16 [4][2048][1024]   (K-major)
  //   [+40MB, +56MB)     W2T bf16 [4][1024][2048]   (K-major)
  char* w = (char*)d_ws;
  float*  wts = (float*)w;
  __bf16* xbf = (__bf16*)(w + (64 << 10));
  __bf16* H   = (__bf16*)(w + (64 << 10) + (8u << 20));
  __bf16* W1T = (__bf16*)(w + (64 << 10) + (24u << 20));
  __bf16* W2T = (__bf16*)(w + (64 << 10) + (40u << 20));

  // One-time precision/layout prep (memory-bound, ~5us of HBM traffic)
  cvt_fp32_bf16<<<(TOKENS * DIM) / (4 * 256), 256, 0, stream>>>(x, xbf);
  transpose_cvt<<<dim3(HID / 32, DIM / 32, NEXP), 256, 0, stream>>>(W1, W1T, DIM, HID);
  transpose_cvt<<<dim3(DIM / 32, HID / 32, NEXP), 256, 0, stream>>>(W2, W2T, HID, DIM);

  moe_router<<<TOKENS / 8, 256, 0, stream>>>(x, Wr, br, wts);

  for (int e = 0; e < NEXP; ++e) {
    moe_gemm<DIM, 0><<<dim3(TOKENS / 128, HID / 128), 256, 0, stream>>>(
        xbf, W1T + (size_t)e * HID * DIM, b1 + (size_t)e * HID,
        nullptr, nullptr, H, e);
    moe_gemm<HID, 1><<<dim3(TOKENS / 128, DIM / 128), 256, 0, stream>>>(
        H, W2T + (size_t)e * DIM * HID, b2 + (size_t)e * DIM,
        wts, out, nullptr, e);
  }
}
